// QuantumLayer_70574902607968
// MI455X (gfx1250) — matrix operand order repaired and checked by reading, not compiled, not run on hardware
//
#include <hip/hip_runtime.h>
#include <hip/hip_bf16.h>
#include <math.h>

// ---------------- problem dims ----------------
#define BB 8
#define SS 512
#define DD 1024
#define PP 128
#define HH 8
#define MM (BB*SS)        // 4096 rows
#define TWO_PI 6.28318530717958647692f

typedef __attribute__((ext_vector_type(16))) __bf16 v16bf;
typedef __attribute__((ext_vector_type(8)))  __bf16 v8bf;
typedef __attribute__((ext_vector_type(8)))  float  v8f;

// ---------------- CDNA5 async global->LDS copy (ASYNCcnt path) ----------------
// GLOBAL_LOAD_ASYNC_TO_LDS_B128: VDST = per-lane LDS byte address (LDS aperture
// puts the byte offset in addr[31:0]), VADDR = 64-bit global address, GV mode.
__device__ __forceinline__ void async_copy16(const void* gsrc, void* ldst) {
    unsigned lds = (unsigned)(unsigned long long)ldst;
    unsigned long long ga = (unsigned long long)gsrc;
    asm volatile("global_load_async_to_lds_b128 %0, %1, off"
                 :: "v"(lds), "v"(ga) : "memory");
}
#if defined(__has_builtin)
# if __has_builtin(__builtin_amdgcn_s_wait_asynccnt)
#  define WAIT_ASYNC0() __builtin_amdgcn_s_wait_asynccnt(0)
# endif
#endif
#ifndef WAIT_ASYNC0
# define WAIT_ASYNC0() asm volatile("s_wait_asynccnt 0" ::: "memory")
#endif

// ---------------- helpers ----------------
__device__ __forceinline__ __bf16 f2bf(float f) {
    union { float f; unsigned u; } v; v.f = f;
    unsigned r = v.u + 0x7FFFu + ((v.u >> 16) & 1u);   // round-to-nearest-even
    unsigned short h = (unsigned short)(r >> 16);
    return __builtin_bit_cast(__bf16, h);
}
__device__ __forceinline__ float sigmoidf_(float x) { return 1.0f / (1.0f + expf(-x)); }

// A fragment (16x32 bf16 tile, row-major, leading dim lda): per CDNA5 ISA 7.12.2
// lane ln holds row M=ln; elements 0..7 = K[half*8 .. half*8+7],
// elements 8..15 = K[16+half*8 .. 16+half*8+7]  -> two contiguous 16B runs.
__device__ __forceinline__ v16bf ld_a(const __bf16* __restrict__ A, int lda) {
    const int lane = threadIdx.x & 31;
    const int half = lane >> 4;
    const int ln   = lane & 15;
    const v8bf lo = *(const v8bf*)(A + ln * lda + half * 8);
    const v8bf hi = *(const v8bf*)(A + ln * lda + half * 8 + 16);
    return __builtin_shufflevector(lo, hi, 0,1,2,3,4,5,6,7,8,9,10,11,12,13,14,15);
}

// B fragment from pre-packed weights: 32 contiguous bytes per lane, coalesced.
__device__ __forceinline__ v16bf ld_b(const __bf16* __restrict__ Bp) {
    const int lane = threadIdx.x & 31;
    return *(const v16bf*)(Bp + lane * 16);
}

__device__ __forceinline__ v8f wmma_acc(v16bf a, v16bf b, v8f acc) {
    return __builtin_amdgcn_wmma_f32_16x16x32_bf16(
        false, a, false, b, (short)0, acc, false, false);
}

// ---------------- kernel 0a: fp32 -> bf16 convert (activations) ----------------
__global__ void k_cvt_bf16(const float* __restrict__ src, __bf16* __restrict__ dst, int n) {
    int i = blockIdx.x * blockDim.x + threadIdx.x;
    int stride = gridDim.x * blockDim.x;
    for (; i < n; i += stride) dst[i] = f2bf(src[i]);
}

// ---------------- kernel 0b: pack row-major [K,N] f32 weight into WMMA
// B-fragment order bf16. Packed index: ((kt*nT + nt)*32 + lane)*16 + e, where
// fragment element e of lane (half,ln) is B[kt*32 + (e<8 ? half*8+e : 16+half*8+e-8)][nt*16 + ln].
// blockIdx.y selects an independent matrix (per-head weights).
__global__ void k_pack_b(const float* __restrict__ src, __bf16* __restrict__ dst,
                         int K, int N) {
    const size_t per = (size_t)K * N;
    const float* s = src + (size_t)blockIdx.y * per;
    __bf16*      d = dst + (size_t)blockIdx.y * per;
    const int nT = N >> 4;
    const int total = K * N;
    for (int idx = blockIdx.x * blockDim.x + threadIdx.x; idx < total;
         idx += gridDim.x * blockDim.x) {
        int e    = idx & 15;
        int lane = (idx >> 4) & 31;
        int t    = idx >> 9;
        int nt   = t % nT, kt = t / nT;
        int half = lane >> 4, ln = lane & 15;
        int k = kt * 32 + (e < 8 ? half * 8 + e : 16 + half * 8 + (e - 8));
        int n = nt * 16 + ln;
        d[idx] = f2bf(s[(size_t)k * N + n]);
    }
}

// ---------------- kernel 1: encode (amp/phase GEMMs) + real + LayerNorm1 ----------------
// grid = 256 (M tiles of 16), block = 256 (8 waves; wave w owns N-tile w of 8)
__global__ void k_encode(const __bf16* __restrict__ xb,
                         const __bf16* __restrict__ WabP,
                         const __bf16* __restrict__ WpbP,
                         const float* __restrict__ ba,  const float* __restrict__ bp,
                         const float* __restrict__ g1,  const float* __restrict__ bt1,
                         float* __restrict__ realb,     __bf16* __restrict__ xnb)
{
    __shared__ __bf16 As[16 * DD];          // 32 KB
    __shared__ float  lr[16][PP];           // 8 KB
    __shared__ float  mu[16], rs[16];
    const int mt   = blockIdx.x;
    const int wave = threadIdx.x >> 5;
    const int lane = threadIdx.x & 31;
    const int half = lane >> 4;
    const int ln   = lane & 15;
    const int n0   = wave * 16;

    // async, coalesced A stage: 16 rows x 1024 cols bf16 (16B per lane per op)
    {
        const __bf16* g = xb + (size_t)mt * 16 * DD;
        for (int i = threadIdx.x; i < (16 * DD) / 8; i += 256)
            async_copy16(g + (size_t)i * 8, As + (size_t)i * 8);
        WAIT_ASYNC0();
    }
    __syncthreads();

    v8f aA = {}, aP = {};
    for (int kt = 0; kt < DD / 32; ++kt) {
        v16bf af = ld_a(As + kt * 32, DD);
        v16bf bA = ld_b(WabP + ((size_t)(kt * 8 + wave) << 9));
        v16bf bP = ld_b(WpbP + ((size_t)(kt * 8 + wave) << 9));
        aA = wmma_acc(af, bA, aA);
        aP = wmma_acc(af, bP, aP);
    }
#pragma unroll
    for (int r = 0; r < 8; ++r) {
        int m = r + 8 * half;
        int p = n0 + ln;
        float amp = tanhf(aA[r] + ba[p]);
        float ph  = sigmoidf_(aP[r] + bp[p]) * TWO_PI;
        float re  = amp * cosf(ph);
        lr[m][p] = re;
        realb[((size_t)mt * 16 + m) * PP + p] = re;
    }
    __syncthreads();
    if (threadIdx.x < 16) {
        int m = threadIdx.x;
        float s = 0.f;
        for (int p = 0; p < PP; ++p) s += lr[m][p];
        float mean = s * (1.0f / PP);
        float var = 0.f;
        for (int p = 0; p < PP; ++p) { float d = lr[m][p] - mean; var += d * d; }
        mu[m] = mean;
        rs[m] = rsqrtf(var * (1.0f / PP) + 1e-5f);
    }
    __syncthreads();
#pragma unroll
    for (int e = 0; e < 8; ++e) {
        int idx = threadIdx.x * 8 + e;
        int m = idx >> 7, p = idx & 127;
        float xv = (lr[m][p] - mu[m]) * rs[m] * g1[p] + bt1[p];
        xnb[((size_t)mt * 16 + m) * PP + p] = f2bf(xv);
    }
}

// ---------------- kernel 2: per-head t1 = gelu(xn @ W1[h] + bh1[h]) ----------------
// grid = (256, H), block = 256 (8 waves x 2 N-tiles; N=256 -> nT=16)
__global__ void k_mlp1(const __bf16* __restrict__ xnb, const __bf16* __restrict__ W1bP,
                       const float* __restrict__ bh1, __bf16* __restrict__ t1b)
{
    __shared__ __bf16 As[16 * PP];          // 4 KB
    const int mt   = blockIdx.x;
    const int h    = blockIdx.y;
    const int wave = threadIdx.x >> 5;
    const int lane = threadIdx.x & 31;
    const int half = lane >> 4, ln = lane & 15;
    const __bf16* Bh = W1bP + (size_t)h * PP * 256;
    {
        const __bf16* g = xnb + (size_t)mt * 16 * PP;
        for (int i = threadIdx.x; i < (16 * PP) / 8; i += 256)
            async_copy16(g + (size_t)i * 8, As + (size_t)i * 8);
        WAIT_ASYNC0();
    }
    __syncthreads();
#pragma unroll
    for (int tt = 0; tt < 2; ++tt) {
        int nt = wave + tt * 8;             // 0..15
        v8f acc = {};
#pragma unroll
        for (int kt = 0; kt < PP / 32; ++kt) {
            v16bf af = ld_a(As + kt * 32, PP);
            v16bf bf = ld_b(Bh + ((size_t)(kt * 16 + nt) << 9));
            acc = wmma_acc(af, bf, acc);
        }
#pragma unroll
        for (int r = 0; r < 8; ++r) {
            int m = r + 8 * half;
            int n = nt * 16 + ln;
            float v = acc[r] + bh1[h * 256 + n];
            float gl = 0.5f * v * (1.0f + erff(v * 0.70710678118654752f)); // exact gelu
            t1b[((size_t)h * MM + mt * 16 + m) * 256 + n] = f2bf(gl);
        }
    }
}

// ---------------- kernel 3: per-head t = t1 @ W2[h] + bh2[h] (fp32 out) ----------------
// grid = (256, H), block = 256 (8 waves x 1 N-tile; N=128 -> nT=8)
__global__ void k_mlp2(const __bf16* __restrict__ t1b, const __bf16* __restrict__ W2bP,
                       const float* __restrict__ bh2, float* __restrict__ tbuf)
{
    __shared__ __bf16 As[16 * 256];         // 8 KB
    const int mt   = blockIdx.x;
    const int h    = blockIdx.y;
    const int wave = threadIdx.x >> 5;
    const int lane = threadIdx.x & 31;
    const int half = lane >> 4, ln = lane & 15;
    const __bf16* Bh = W2bP + (size_t)h * 256 * PP;
    {
        const __bf16* g = t1b + ((size_t)h * MM + mt * 16) * 256;
        for (int i = threadIdx.x; i < (16 * 256) / 8; i += 256)
            async_copy16(g + (size_t)i * 8, As + (size_t)i * 8);
        WAIT_ASYNC0();
    }
    __syncthreads();
    v8f acc = {};
#pragma unroll
    for (int kt = 0; kt < 256 / 32; ++kt) {
        v16bf af = ld_a(As + kt * 32, 256);
        v16bf bf = ld_b(Bh + ((size_t)(kt * 8 + wave) << 9));
        acc = wmma_acc(af, bf, acc);
    }
#pragma unroll
    for (int r = 0; r < 8; ++r) {
        int m = r + 8 * half;
        int p = wave * 16 + ln;
        tbuf[((size_t)h * MM + mt * 16 + m) * PP + p] = acc[r] + bh2[h * PP + p];
    }
}

// ---------------- kernel 4: per-channel autocorrelation + sigmoid gate ----------------
// grid = H*B*P = 8192 channels, block = 256. Signal lives in LDS; each thread 2 lags.
__global__ void k_corr_gate(const float* __restrict__ tbuf, const float* __restrict__ realb,
                            float* __restrict__ comb)
{
    __shared__ float sig[SS];
    const int blk = blockIdx.x;
    const int h   = blk >> 10;          // / (B*P)
    const int rem = blk & 1023;
    const int b   = rem >> 7;
    const int p   = rem & 127;
    const float* base = tbuf + ((size_t)h * MM + b * SS) * PP + p;
    for (int i = threadIdx.x; i < SS; i += blockDim.x) sig[i] = base[(size_t)i * PP];
    __syncthreads();
    for (int j = threadIdx.x; j < SS; j += blockDim.x) {
        int d   = j - SS / 2;           // lag in [-256, 255]
        int klo = d < 0 ? -d : 0;
        int khi = d > 0 ? SS - d : SS;
        float s = 0.f;
        for (int k = klo; k < khi; ++k) s += sig[k] * sig[k + d];
        float g = realb[((size_t)b * SS + j) * PP + p] * sigmoidf_(s);
        comb[(((size_t)b * SS + j) * HH + h) * PP + p] = g;
    }
}

// ---------------- kernel 5: LayerNorm2 over P per (b,s,h) -> bf16 ----------------
__global__ void k_norm2(const float* __restrict__ comb, const float* __restrict__ g2,
                        const float* __restrict__ bt2, __bf16* __restrict__ nrmb)
{
    __shared__ float red[PP];
    __shared__ float smu, srs;
    const size_t row = blockIdx.x;
    const int p = threadIdx.x;
    float v = comb[row * PP + p];
    red[p] = v; __syncthreads();
    for (int s = 64; s > 0; s >>= 1) { if (p < s) red[p] += red[p + s]; __syncthreads(); }
    if (p == 0) smu = red[0] * (1.0f / PP);
    __syncthreads();
    float d = v - smu;
    red[p] = d * d; __syncthreads();
    for (int s = 64; s > 0; s >>= 1) { if (p < s) red[p] += red[p + s]; __syncthreads(); }
    if (p == 0) srs = rsqrtf(red[0] * (1.0f / PP) + 1e-5f);
    __syncthreads();
    nrmb[row * PP + p] = f2bf(d * srs * g2[p] + bt2[p]);
}

// ---------------- kernel 6: out = nrm @ Wo + bo + x ----------------
// grid = (128 M-blocks of 32 rows, 8 N-groups of 128), block = 256.
// Each wave: one B fragment per K-step feeds TWO WMMAs (rows 0-15 / 16-31).
__global__ void k_out(const __bf16* __restrict__ nrmb, const __bf16* __restrict__ WobP,
                      const float* __restrict__ bo, const float* __restrict__ x,
                      float* __restrict__ out)
{
    __shared__ __bf16 As[32 * DD];          // 64 KB
    const int mt   = blockIdx.x;            // 0..127 (32-row blocks)
    const int ng   = blockIdx.y;            // 0..7
    const int wave = threadIdx.x >> 5;
    const int lane = threadIdx.x & 31;
    const int half = lane >> 4, ln = lane & 15;
    const int nt   = ng * 8 + wave;         // global N-tile 0..63 (nT = 64)
    {
        const __bf16* g = nrmb + (size_t)mt * 32 * DD;
        for (int i = threadIdx.x; i < (32 * DD) / 8; i += 256)
            async_copy16(g + (size_t)i * 8, As + (size_t)i * 8);
        WAIT_ASYNC0();
    }
    __syncthreads();
    v8f acc0 = {}, acc1 = {};
    for (int kt = 0; kt < DD / 32; ++kt) {
        v16bf bf = ld_b(WobP + ((size_t)(kt * 64 + nt) << 9));
        v16bf a0 = ld_a(As + kt * 32, DD);
        v16bf a1 = ld_a(As + 16 * DD + kt * 32, DD);
        acc0 = wmma_acc(a0, bf, acc0);
        acc1 = wmma_acc(a1, bf, acc1);
    }
#pragma unroll
    for (int r = 0; r < 8; ++r) {
        int m = r + 8 * half;
        int n = nt * 16 + ln;
        size_t i0 = ((size_t)mt * 32 + m) * DD + n;
        size_t i1 = ((size_t)mt * 32 + 16 + m) * DD + n;
        out[i0] = acc0[r] + bo[n] + x[i0];
        out[i1] = acc1[r] + bo[n] + x[i1];
    }
}

// ---------------- workspace layout (bytes) ----------------
static constexpr size_t OFF_XB   = 0;                                    // x bf16   [4096,1024]
static constexpr size_t OFF_WAB  = OFF_XB   + (size_t)MM * DD * 2;       // Wa packed bf16
static constexpr size_t OFF_WPB  = OFF_WAB  + (size_t)DD * PP * 2;       // Wp packed
static constexpr size_t OFF_W1B  = OFF_WPB  + (size_t)DD * PP * 2;       // W1 packed [8][128*256]
static constexpr size_t OFF_W2B  = OFF_W1B  + (size_t)HH * PP * 256 * 2; // W2 packed [8][256*128]
static constexpr size_t OFF_WOB  = OFF_W2B  + (size_t)HH * 256 * PP * 2; // Wo packed [1024*1024]
static constexpr size_t OFF_REAL = OFF_WOB  + (size_t)DD * DD * 2;       // real f32 [4096,128]
static constexpr size_t OFF_XNB  = OFF_REAL + (size_t)MM * PP * 4;       // xn bf16  [4096,128]
static constexpr size_t OFF_T1B  = OFF_XNB  + (size_t)MM * PP * 2;       // t1 bf16  [8,4096,256]
static constexpr size_t OFF_T    = OFF_T1B  + (size_t)HH * MM * 256 * 2; // t  f32   [8,4096,128]
static constexpr size_t OFF_COMB = OFF_T    + (size_t)HH * MM * PP * 4;  // comb f32 [4096,8,128]
static constexpr size_t OFF_NRM  = OFF_COMB + (size_t)MM * HH * PP * 4;  // nrm bf16 [4096,1024]

extern "C" void kernel_launch(void* const* d_in, const int* in_sizes, int n_in,
                              void* d_out, int out_size, void* d_ws, size_t ws_size,
                              hipStream_t stream) {
    const float* x   = (const float*)d_in[0];
    const float* Wa  = (const float*)d_in[1];
    const float* ba  = (const float*)d_in[2];
    const float* Wp  = (const float*)d_in[3];
    const float* bp  = (const float*)d_in[4];
    const float* g1  = (const float*)d_in[5];
    const float* bt1 = (const float*)d_in[6];
    const float* W1  = (const float*)d_in[7];
    const float* bh1 = (const float*)d_in[8];
    const float* W2  = (const float*)d_in[9];
    const float* bh2 = (const float*)d_in[10];
    const float* g2  = (const float*)d_in[11];
    const float* bt2 = (const float*)d_in[12];
    const float* Wo  = (const float*)d_in[13];
    const float* bo  = (const float*)d_in[14];
    float* out = (float*)d_out;

    char* ws = (char*)d_ws;
    __bf16* xb   = (__bf16*)(ws + OFF_XB);
    __bf16* WabP = (__bf16*)(ws + OFF_WAB);
    __bf16* WpbP = (__bf16*)(ws + OFF_WPB);
    __bf16* W1bP = (__bf16*)(ws + OFF_W1B);
    __bf16* W2bP = (__bf16*)(ws + OFF_W2B);
    __bf16* WobP = (__bf16*)(ws + OFF_WOB);
    float*  realb= (float*)(ws + OFF_REAL);
    __bf16* xnb  = (__bf16*)(ws + OFF_XNB);
    __bf16* t1b  = (__bf16*)(ws + OFF_T1B);
    float*  tbuf = (float*)(ws + OFF_T);
    float*  comb = (float*)(ws + OFF_COMB);
    __bf16* nrmb = (__bf16*)(ws + OFF_NRM);

    // 0) convert activations; pack weights into WMMA B-fragment order
    k_cvt_bf16<<<2048, 256, 0, stream>>>(x, xb, MM * DD);
    k_pack_b<<<dim3(256, 1), 256, 0, stream>>>(Wa, WabP, DD, PP);
    k_pack_b<<<dim3(256, 1), 256, 0, stream>>>(Wp, WpbP, DD, PP);
    k_pack_b<<<dim3(64, HH), 256, 0, stream>>>(W1, W1bP, PP, 256);
    k_pack_b<<<dim3(64, HH), 256, 0, stream>>>(W2, W2bP, 256, PP);
    k_pack_b<<<dim3(1024, 1), 256, 0, stream>>>(Wo, WobP, DD, DD);

    // 1) encode: amp/phase WMMA GEMMs + real + LayerNorm1
    k_encode<<<MM / 16, 256, 0, stream>>>(xb, WabP, WpbP, ba, bp, g1, bt1, realb, xnb);

    // 2) per-head MLP stage 1 (WMMA + gelu)
    k_mlp1<<<dim3(MM / 16, HH), 256, 0, stream>>>(xnb, W1bP, bh1, t1b);

    // 3) per-head MLP stage 2 (WMMA)
    k_mlp2<<<dim3(MM / 16, HH), 256, 0, stream>>>(t1b, W2bP, bh2, tbuf);

    // 4) autocorrelation + sigmoid gate
    k_corr_gate<<<HH * BB * PP, 256, 0, stream>>>(tbuf, realb, comb);

    // 5) LayerNorm2
    k_norm2<<<MM * HH, PP, 0, stream>>>(comb, g2, bt2, nrmb);

    // 6) output GEMM + bias + residual (2 M-tiles per wave)
    k_out<<<dim3(MM / 32, 8), 256, 0, stream>>>(nrmb, WobP, bo, x, out);
}
